// LinearAttentionBlock_53317724013008
// MI455X (gfx1250) — matrix operand order repaired and checked by reading, not compiled, not run on hardware
//
#include <hip/hip_runtime.h>

// ---------------------------------------------------------------------------
// Problem constants (B=1, S=2048, D=1024)
// ---------------------------------------------------------------------------
#define SEQ      2048
#define DIM      1024
#define KEY_DIM  512
#define VAL_DIM  1024
#define CONV_DIM 2048          // 2*KEY_DIM + VAL_DIM
#define HK       8
#define HV       16
#define DK       64
#define DV       64
#define NE       8
#define IMOE     512
#define KW       4

typedef __attribute__((ext_vector_type(16))) __bf16        v16bf;
typedef __attribute__((ext_vector_type(8)))  float         v8f;
typedef __attribute__((ext_vector_type(4)))  unsigned int  u32x4;

union FragBF { u32x4 q[2]; v16bf v; };

static __device__ __forceinline__ unsigned short f2bf(float f) {
    unsigned int u = __float_as_uint(f);
    unsigned int r = (u + 0x7FFFu + ((u >> 16) & 1u)) >> 16;
    return (unsigned short)r;
}
static __device__ __forceinline__ float bf2f(unsigned short b) {
    return __uint_as_float(((unsigned int)b) << 16);
}
static __device__ __forceinline__ float silu(float x) {
    return x / (1.0f + __expf(-x));
}

// ---------------------------------------------------------------------------
// Weight preparation: f32 -> bf16 (elementwise), and LDS-tiled transpose+convert
// so every GEMM B operand is bf16 row-major [N][K].
// ---------------------------------------------------------------------------
__global__ __launch_bounds__(256)
void convert_bf16_kernel(const float* __restrict__ in,
                         unsigned short* __restrict__ out, int n)
{
    int i = blockIdx.x * 256 + threadIdx.x;
    if (i < n) out[i] = f2bf(in[i]);
}

// in: f32 [K][N]  ->  out: bf16 [N][K]   (K%32==0, N%32==0)
__global__ __launch_bounds__(256)
void transpose_convert_kernel(const float* __restrict__ in,
                              unsigned short* __restrict__ out, int K, int N)
{
    __shared__ float tile[32][33];
    const int n0 = blockIdx.x * 32;
    const int k0 = blockIdx.y * 32;
    const int tx = threadIdx.x & 31;
    const int ty = threadIdx.x >> 5;   // 0..7
    #pragma unroll
    for (int j = 0; j < 4; ++j)
        tile[ty + j * 8][tx] = in[(size_t)(k0 + ty + j * 8) * N + n0 + tx];
    __syncthreads();
    #pragma unroll
    for (int j = 0; j < 4; ++j)
        out[(size_t)(n0 + ty + j * 8) * K + k0 + tx] = f2bf(tile[tx][ty + j * 8]);
}

// ---------------------------------------------------------------------------
// bf16 WMMA GEMM:  C[M,N] (f32) = A[M,K] (bf16) * B[N,K] (bf16)
// Block tile 64(M) x 128(N), BK=32; 8 waves, each wave owns a 32x32 output
// (2 A-frags x 2 B-frags -> 4 v_wmma per K-step).
//   rowScale : optional per-row (token) scale with stride rsStride
//   accum    : 1 -> C += result, 0 -> C = result
// Requires M%64==0, N%128==0, K%32==0 (true for every call here).
// ---------------------------------------------------------------------------
#define BK 32
#define ASTR 48   // LDS row stride (bf16 elems); 96B rows keep 16B chunk alignment

__global__ __launch_bounds__(256)
void gemm_wmma_kernel(const unsigned short* __restrict__ A,
                      const unsigned short* __restrict__ B,
                      float* __restrict__ C,
                      int M, int N, int K,
                      const float* __restrict__ rowScale, int rsStride,
                      int accum)
{
    __shared__ unsigned short As[64 * ASTR];
    __shared__ unsigned short Bs[128 * ASTR];

    const int tid  = threadIdx.x;
    const int lane = tid & 31;
    const int wave = tid >> 5;
    const int wm   = wave >> 2;     // 0..1  (M groups of 32)
    const int wn   = wave & 3;      // 0..3  (N groups of 32)
    const int m0   = blockIdx.y * 64;
    const int n0   = blockIdx.x * 128;
    const int laneHalf = lane >> 4; // 0/1
    const int laneMod  = lane & 15;

    const v8f vzero = {0.f,0.f,0.f,0.f,0.f,0.f,0.f,0.f};
    v8f acc[2][2] = {{vzero, vzero}, {vzero, vzero}};

    const int ldRow = tid >> 2;           // 0..63
    const int ldCol = (tid & 3) * 8;      // 0,8,16,24

    for (int k0 = 0; k0 < K; k0 += BK) {
        // ---- cooperative tile loads: pure 16B bf16 copies, no conversion ----
        const unsigned short* srcA  = A + (size_t)(m0 + ldRow) * K + k0 + ldCol;
        const unsigned short* srcB0 = B + (size_t)(n0 + ldRow) * K + k0 + ldCol;
        const unsigned short* srcB1 = B + (size_t)(n0 + 64 + ldRow) * K + k0 + ldCol;
        *(u32x4*)(&As[ldRow * ASTR + ldCol])        = *(const u32x4*)srcA;
        *(u32x4*)(&Bs[ldRow * ASTR + ldCol])        = *(const u32x4*)srcB0;
        *(u32x4*)(&Bs[(64 + ldRow) * ASTR + ldCol]) = *(const u32x4*)srcB1;
        if (k0 + BK < K) {
            // locality=3 -> nearest-cache (WGP-scope) prefetch of next tiles
            __builtin_prefetch(srcA + BK, 0, 3);    // global_prefetch_b8
            __builtin_prefetch(srcB0 + BK, 0, 3);
            __builtin_prefetch(srcB1 + BK, 0, 3);
        }
        __syncthreads();

        // ---- fragments per ISA 7.12.2 layouts ----
        FragBF fa[2], fb[2];
        #pragma unroll
        for (int s = 0; s < 2; ++s) {
            const int aRow = wm * 32 + s * 16 + laneMod;
            fa[s].q[0] = *(const u32x4*)(&As[aRow * ASTR + laneHalf * 8]);
            fa[s].q[1] = *(const u32x4*)(&As[aRow * ASTR + 16 + laneHalf * 8]);
            const int bRow = wn * 32 + s * 16 + laneMod;
            fb[s].q[0] = *(const u32x4*)(&Bs[bRow * ASTR + laneHalf * 8]);
            fb[s].q[1] = *(const u32x4*)(&Bs[bRow * ASTR + 16 + laneHalf * 8]);
        }
        acc[0][0] = __builtin_amdgcn_wmma_f32_16x16x32_bf16(false, fa[0].v, false, fb[0].v,
                                                            (short)0, acc[0][0], false, false);
        acc[0][1] = __builtin_amdgcn_wmma_f32_16x16x32_bf16(false, fa[0].v, false, fb[1].v,
                                                            (short)0, acc[0][1], false, false);
        acc[1][0] = __builtin_amdgcn_wmma_f32_16x16x32_bf16(false, fa[1].v, false, fb[0].v,
                                                            (short)0, acc[1][0], false, false);
        acc[1][1] = __builtin_amdgcn_wmma_f32_16x16x32_bf16(false, fa[1].v, false, fb[1].v,
                                                            (short)0, acc[1][1], false, false);
        __syncthreads();
    }

    // ---- epilogue: C/D layout lanes 0-15 M=v, lanes 16-31 M=8+v ----
    #pragma unroll
    for (int sm = 0; sm < 2; ++sm) {
        #pragma unroll
        for (int sn = 0; sn < 2; ++sn) {
            v8f a = acc[sm][sn];
            const int rb  = m0 + wm * 32 + sm * 16 + laneHalf * 8;
            const int col = n0 + wn * 32 + sn * 16 + laneMod;
            #pragma unroll
            for (int v = 0; v < 8; ++v) {
                int row = rb + v;
                float val = a[v];
                if (rowScale) val *= rowScale[(size_t)row * rsStride];
                size_t idx = (size_t)row * N + col;
                if (accum) C[idx] += val; else C[idx] = val;
            }
        }
    }
}

// ---------------------------------------------------------------------------
// RMS norm (f32 in) -> bf16 out, weight applied as (1 + w)
// ---------------------------------------------------------------------------
__global__ __launch_bounds__(256)
void rmsnorm_bf16_kernel(const float* __restrict__ x, const float* __restrict__ w,
                         unsigned short* __restrict__ out, int Dd)
{
    const int row = blockIdx.x;
    const float* xr = x + (size_t)row * Dd;
    __shared__ float red[8];
    float s = 0.f;
    for (int i = threadIdx.x; i < Dd; i += 256) { float v = xr[i]; s += v * v; }
    for (int o = 16; o > 0; o >>= 1) s += __shfl_xor(s, o, 32);
    if ((threadIdx.x & 31) == 0) red[threadIdx.x >> 5] = s;
    __syncthreads();
    float tot = 0.f;
    #pragma unroll
    for (int i = 0; i < 8; ++i) tot += red[i];
    float r = rsqrtf(tot / (float)Dd + 1e-6f);
    for (int i = threadIdx.x; i < Dd; i += 256)
        out[(size_t)row * Dd + i] = f2bf(xr[i] * r * (1.f + w[i]));
}

// ---------------------------------------------------------------------------
// a/b projections (N=16 each) + beta = sigmoid(b),
// decay = exp(-exp(A_log)*softplus(a+dt_bias))
// ---------------------------------------------------------------------------
__global__ __launch_bounds__(256)
void ab_kernel(const unsigned short* __restrict__ hbf,
               const float* __restrict__ Wa, const float* __restrict__ Wb,
               const float* __restrict__ dt_bias, const float* __restrict__ A_log,
               float* __restrict__ decay, float* __restrict__ beta)
{
    const int s = blockIdx.x;
    const int lane = threadIdx.x & 31, wave = threadIdx.x >> 5;
    const unsigned short* hr = hbf + (size_t)s * DIM;
    __shared__ float av[HV], bv[HV];
    for (int j = 0; j < 4; ++j) {
        int o = wave * 4 + j;                 // 0..31: a[0..15], b[0..15]
        const float* W = (o < HV) ? Wa : Wb;
        int col = o & 15;
        float acc = 0.f;
        for (int d = lane; d < DIM; d += 32) acc += bf2f(hr[d]) * W[d * HV + col];
        for (int off = 16; off > 0; off >>= 1) acc += __shfl_xor(acc, off, 32);
        if (lane == 0) { if (o < HV) av[col] = acc; else bv[col] = acc; }
    }
    __syncthreads();
    if (threadIdx.x < HV) {
        int h = threadIdx.x;
        beta[s * HV + h] = 1.f / (1.f + __expf(-bv[h]));
        float aa = av[h] + dt_bias[h];
        float sp = (aa > 20.f) ? aa : log1pf(__expf(aa));
        decay[s * HV + h] = __expf(-__expf(A_log[h]) * sp);
    }
}

// ---------------------------------------------------------------------------
// Depthwise causal conv1d (KW=4) + SiLU; also writes new_conv_state
// ---------------------------------------------------------------------------
__global__ __launch_bounds__(256)
void conv_silu_kernel(const float* __restrict__ qkv,
                      const float* __restrict__ conv_state,
                      const float* __restrict__ w,
                      float* __restrict__ out,
                      float* __restrict__ new_conv_state)
{
    const int c = blockIdx.x * 256 + threadIdx.x;  // channel 0..2047
    const int s = blockIdx.y;
    float acc = 0.f;
    #pragma unroll
    for (int j = 0; j < KW; ++j) {
        int t = s + j - (KW - 1);
        float v = (t < 0) ? conv_state[(size_t)(t + KW - 1) * CONV_DIM + c]
                          : qkv[(size_t)t * CONV_DIM + c];
        acc += w[c * KW + j] * v;
    }
    out[(size_t)s * CONV_DIM + c] = silu(acc);
    if (s >= SEQ - (KW - 1))
        new_conv_state[(size_t)(s - (SEQ - (KW - 1))) * CONV_DIM + c] =
            qkv[(size_t)s * CONV_DIM + c];
}

// ---------------------------------------------------------------------------
// In-place per-head l2norm of q (cols 0..511, also * DK^-0.5) and k (512..1023)
// ---------------------------------------------------------------------------
__global__ __launch_bounds__(64)
void l2norm_kernel(float* __restrict__ qk)
{
    const int g = blockIdx.x;   // 0..15 (q heads 0..7, k heads 8..15)
    const int s = blockIdx.y;
    const int v = threadIdx.x;
    float* p = qk + (size_t)s * CONV_DIM + g * DK;
    __shared__ float red[2];
    float x = p[v];
    float ss = x * x;
    for (int o = 16; o > 0; o >>= 1) ss += __shfl_xor(ss, o, 32);
    if ((v & 31) == 0) red[v >> 5] = ss;
    __syncthreads();
    float r = rsqrtf(red[0] + red[1] + 1e-6f);
    if (g < HK) r *= 0.125f;   // DK^-0.5
    p[v] = x * r;
}

// ---------------------------------------------------------------------------
// Gated delta-rule recurrent scan: one block per value head, 64x64 state
// held in registers (one column per lane).
// ---------------------------------------------------------------------------
__global__ __launch_bounds__(64)
void scan_kernel(const float* __restrict__ qkv,
                 const float* __restrict__ decay,
                 const float* __restrict__ beta,
                 const float* __restrict__ state_in,
                 float* __restrict__ o,
                 float* __restrict__ state_out)
{
    const int h  = blockIdx.x;     // 0..15
    const int v  = threadIdx.x;    // state column
    const int kh = h >> 1;         // source q/k head
    float st[DK];
    #pragma unroll
    for (int k = 0; k < DK; ++k)
        st[k] = state_in[(size_t)h * (DK * DV) + k * DV + v];

    __shared__ float sk[DK], sq[DK];
    for (int s = 0; s < SEQ; ++s) {
        const float* row = qkv + (size_t)s * CONV_DIM;
        sq[v] = row[kh * DK + v];
        sk[v] = row[KEY_DIM + kh * DK + v];
        __syncthreads();
        float vt = row[2 * KEY_DIM + h * DV + v];
        float d  = decay[s * HV + h];
        float bt = beta[s * HV + h];
        float mem = 0.f;
        #pragma unroll
        for (int k = 0; k < DK; ++k) { st[k] *= d; mem += sk[k] * st[k]; }
        float delta = (vt - mem) * bt;
        float ov = 0.f;
        #pragma unroll
        for (int k = 0; k < DK; ++k) { st[k] += sk[k] * delta; ov += sq[k] * st[k]; }
        o[(size_t)s * VAL_DIM + h * DV + v] = ov;
        __syncthreads();
    }
    #pragma unroll
    for (int k = 0; k < DK; ++k)
        state_out[(size_t)h * (DK * DV) + k * DV + v] = st[k];
}

// ---------------------------------------------------------------------------
// Per-head RMS norm of o, * (1+nw), * silu(z) -> bf16 (A operand of out_proj)
// ---------------------------------------------------------------------------
__global__ __launch_bounds__(64)
void gatenorm_kernel(const float* __restrict__ o, const float* __restrict__ z,
                     const float* __restrict__ nw, unsigned short* __restrict__ out)
{
    const int h = blockIdx.x, s = blockIdx.y, v = threadIdx.x;
    size_t idx = (size_t)s * VAL_DIM + h * DV + v;
    float x = o[idx];
    __shared__ float red[2];
    float ss = x * x;
    for (int off = 16; off > 0; off >>= 1) ss += __shfl_xor(ss, off, 32);
    if ((v & 31) == 0) red[v >> 5] = ss;
    __syncthreads();
    float r = rsqrtf((red[0] + red[1]) / (float)DV + 1e-6f);
    out[idx] = f2bf(x * r * (1.f + nw[h * DV + v]) * silu(z[idx]));
}

// ---------------------------------------------------------------------------
// Router: logits (8 dots), softmax, top-2 renormalized -> wdense[S][8];
// plus shared-expert sigmoid gate seg[S].
// ---------------------------------------------------------------------------
__global__ __launch_bounds__(256)
void router_kernel(const unsigned short* __restrict__ tbf,
                   const float* __restrict__ rw,
                   const float* __restrict__ seg_w,
                   float* __restrict__ wdense, float* __restrict__ seg)
{
    const int s = blockIdx.x;
    const int lane = threadIdx.x & 31, wave = threadIdx.x >> 5;
    const unsigned short* tr = tbf + (size_t)s * DIM;
    __shared__ float logits[NE], segred[8];

    float acc = 0.f;
    for (int d = lane; d < DIM; d += 32) acc += bf2f(tr[d]) * rw[wave * DIM + d];
    for (int off = 16; off > 0; off >>= 1) acc += __shfl_xor(acc, off, 32);
    if (lane == 0) logits[wave] = acc;

    float sa = 0.f;
    for (int d = threadIdx.x; d < DIM; d += 256) sa += bf2f(tr[d]) * seg_w[d];
    for (int off = 16; off > 0; off >>= 1) sa += __shfl_xor(sa, off, 32);
    if (lane == 0) segred[wave] = sa;
    __syncthreads();

    if (threadIdx.x == 0) {
        float mx = logits[0];
        for (int e = 1; e < NE; ++e) mx = fmaxf(mx, logits[e]);
        float pe[NE], sum = 0.f;
        for (int e = 0; e < NE; ++e) { pe[e] = __expf(logits[e] - mx); sum += pe[e]; }
        for (int e = 0; e < NE; ++e) pe[e] /= sum;
        int i1 = 0;
        for (int e = 1; e < NE; ++e) if (pe[e] > pe[i1]) i1 = e;
        int i2 = -1;
        for (int e = 0; e < NE; ++e) { if (e == i1) continue; if (i2 < 0 || pe[e] > pe[i2]) i2 = e; }
        float tot = pe[i1] + pe[i2];
        for (int e = 0; e < NE; ++e) wdense[s * NE + e] = 0.f;
        wdense[s * NE + i1] = pe[i1] / tot;
        wdense[s * NE + i2] = pe[i2] / tot;
        float sg = 0.f;
        for (int wv = 0; wv < 8; ++wv) sg += segred[wv];
        seg[s] = 1.f / (1.f + __expf(-sg));
    }
}

// ---------------------------------------------------------------------------
// Shared-expert activation: out = bf16(silu(g) * u), elementwise
// ---------------------------------------------------------------------------
__global__ __launch_bounds__(256)
void silu_mul_kernel(const float* __restrict__ g, const float* __restrict__ u,
                     unsigned short* __restrict__ out, int n)
{
    int i = blockIdx.x * 256 + threadIdx.x;
    if (i < n) out[i] = f2bf(silu(g[i]) * u[i]);
}

// Routed-expert GLU: gu [S][1024] -> act [S][512] bf16
__global__ __launch_bounds__(256)
void glu_kernel(const float* __restrict__ gu, unsigned short* __restrict__ out)
{
    int i = blockIdx.x * 256 + threadIdx.x;   // over SEQ*IMOE
    int s = i >> 9, c = i & (IMOE - 1);
    float gg = gu[(size_t)s * (2 * IMOE) + c];
    float uu = gu[(size_t)s * (2 * IMOE) + IMOE + c];
    out[(size_t)s * IMOE + c] = f2bf(silu(gg) * uu);
}

// ---------------------------------------------------------------------------
// Host-side launcher
// ---------------------------------------------------------------------------
extern "C" void kernel_launch(void* const* d_in, const int* in_sizes, int n_in,
                              void* d_out, int out_size, void* d_ws, size_t ws_size,
                              hipStream_t stream)
{
    const float* x          = (const float*)d_in[0];
    const float* state_in   = (const float*)d_in[1];
    const float* conv_state = (const float*)d_in[2];
    const float* W_qkv      = (const float*)d_in[3];
    const float* W_z        = (const float*)d_in[4];
    const float* W_a        = (const float*)d_in[5];
    const float* W_b        = (const float*)d_in[6];
    const float* conv_w     = (const float*)d_in[7];
    const float* dt_bias    = (const float*)d_in[8];
    const float* A_log      = (const float*)d_in[9];
    const float* norm_w     = (const float*)d_in[10];
    const float* W_out      = (const float*)d_in[11];
    const float* router_w   = (const float*)d_in[12];
    const float* W_gu       = (const float*)d_in[13];  // (E, 2I, D)
    const float* W_down     = (const float*)d_in[14];  // (E, D, I)
    const float* W_sg       = (const float*)d_in[15];
    const float* W_su       = (const float*)d_in[16];
    const float* W_sd       = (const float*)d_in[17];
    const float* W_seg      = (const float*)d_in[18];
    const float* attn_nw    = (const float*)d_in[19];
    const float* ffn_nw     = (const float*)d_in[20];

    // ---------------- workspace layout ----------------
    float* ws       = (float*)d_ws;
    float* qkv_raw  = ws;                                  // S*CONV
    float* qkv_act  = qkv_raw + (size_t)SEQ * CONV_DIM;    // S*CONV
    float* zbuf     = qkv_act + (size_t)SEQ * CONV_DIM;    // S*D
    float* obuf     = zbuf    + (size_t)SEQ * DIM;         // S*D
    float* x1       = obuf    + (size_t)SEQ * DIM;         // S*D
    float* gbuf     = x1      + (size_t)SEQ * DIM;         // S*D
    float* ubuf     = gbuf    + (size_t)SEQ * DIM;         // S*D
    float* decay    = ubuf    + (size_t)SEQ * DIM;         // S*HV
    float* beta     = decay   + (size_t)SEQ * HV;          // S*HV
    float* wdense   = beta    + (size_t)SEQ * HV;          // S*E
    float* seg      = wdense  + (size_t)SEQ * NE;          // S
    unsigned short* hbf = (unsigned short*)(seg + SEQ);    // S*D bf16
    unsigned short* tbf = hbf + (size_t)SEQ * DIM;         // S*D bf16
    unsigned short* abf = tbf + (size_t)SEQ * DIM;         // S*D bf16 (reused)
    // bf16 weight copies, all row-major [N][K]
    unsigned short* wqkv_bf = abf + (size_t)SEQ * DIM;               // CONV x D
    unsigned short* wz_bf   = wqkv_bf + (size_t)CONV_DIM * DIM;      // D x D
    unsigned short* wout_bf = wz_bf   + (size_t)DIM * DIM;           // D x VAL
    unsigned short* wsg_bf  = wout_bf + (size_t)DIM * VAL_DIM;       // VAL x D
    unsigned short* wsu_bf  = wsg_bf  + (size_t)VAL_DIM * DIM;       // VAL x D
    unsigned short* wsd_bf  = wsu_bf  + (size_t)VAL_DIM * DIM;       // D x VAL
    unsigned short* wgu_bf  = wsd_bf  + (size_t)DIM * VAL_DIM;       // E x 2I x D
    unsigned short* wdn_bf  = wgu_bf  + (size_t)NE * 2 * IMOE * DIM; // E x D x I

    float* outX     = (float*)d_out;
    float* outState = outX + (size_t)SEQ * DIM;
    float* outConv  = outState + (size_t)HV * DK * DV;

    // 0) weight prep: transpose+convert [K][N] weights to [N][K] bf16;
    //    expert weights are already [N][K], just convert dtype.
    transpose_convert_kernel<<<dim3(CONV_DIM/32, DIM/32), 256, 0, stream>>>(
        W_qkv, wqkv_bf, DIM, CONV_DIM);
    transpose_convert_kernel<<<dim3(VAL_DIM/32, DIM/32), 256, 0, stream>>>(
        W_z, wz_bf, DIM, VAL_DIM);
    transpose_convert_kernel<<<dim3(DIM/32, VAL_DIM/32), 256, 0, stream>>>(
        W_out, wout_bf, VAL_DIM, DIM);
    transpose_convert_kernel<<<dim3(VAL_DIM/32, DIM/32), 256, 0, stream>>>(
        W_sg, wsg_bf, DIM, VAL_DIM);
    transpose_convert_kernel<<<dim3(VAL_DIM/32, DIM/32), 256, 0, stream>>>(
        W_su, wsu_bf, DIM, VAL_DIM);
    transpose_convert_kernel<<<dim3(DIM/32, VAL_DIM/32), 256, 0, stream>>>(
        W_sd, wsd_bf, VAL_DIM, DIM);
    convert_bf16_kernel<<<(NE * 2 * IMOE * DIM) / 256, 256, 0, stream>>>(
        W_gu, wgu_bf, NE * 2 * IMOE * DIM);
    convert_bf16_kernel<<<(NE * DIM * IMOE) / 256, 256, 0, stream>>>(
        W_down, wdn_bf, NE * DIM * IMOE);

    // 1) attention RMS norm -> hbf
    rmsnorm_bf16_kernel<<<SEQ, 256, 0, stream>>>(x, attn_nw, hbf, DIM);

    // 2) qkv / z projections (WMMA)
    gemm_wmma_kernel<<<dim3(CONV_DIM/128, SEQ/64), 256, 0, stream>>>(
        hbf, wqkv_bf, qkv_raw, SEQ, CONV_DIM, DIM, nullptr, 0, 0);
    gemm_wmma_kernel<<<dim3(VAL_DIM/128, SEQ/64), 256, 0, stream>>>(
        hbf, wz_bf, zbuf, SEQ, VAL_DIM, DIM, nullptr, 0, 0);

    // 3) a/b projections + decay/beta
    ab_kernel<<<SEQ, 256, 0, stream>>>(hbf, W_a, W_b, dt_bias, A_log, decay, beta);

    // 4) conv1d + SiLU (+ new_conv_state)
    conv_silu_kernel<<<dim3(CONV_DIM/256, SEQ), 256, 0, stream>>>(
        qkv_raw, conv_state, conv_w, qkv_act, outConv);

    // 5) l2norm q/k in place (q also scaled by DK^-0.5)
    l2norm_kernel<<<dim3(16, SEQ), 64, 0, stream>>>(qkv_act);

    // 6) recurrent delta-rule scan (+ new_state)
    scan_kernel<<<HV, 64, 0, stream>>>(qkv_act, decay, beta, state_in, obuf, outState);

    // 7) gated group-norm -> abf (bf16)
    gatenorm_kernel<<<dim3(HV, SEQ), 64, 0, stream>>>(obuf, zbuf, norm_w, abf);

    // 8) attn residual: x1 = x; x1 += abf @ out_proj
    hipMemcpyAsync(x1, x, (size_t)SEQ * DIM * sizeof(float),
                   hipMemcpyDeviceToDevice, stream);
    gemm_wmma_kernel<<<dim3(DIM/128, SEQ/64), 256, 0, stream>>>(
        abf, wout_bf, x1, SEQ, DIM, VAL_DIM, nullptr, 0, 1);

    // 9) FFN RMS norm -> tbf
    rmsnorm_bf16_kernel<<<SEQ, 256, 0, stream>>>(x1, ffn_nw, tbf, DIM);

    // 10) router (top-2 weights) + shared-expert sigmoid gate
    router_kernel<<<SEQ, 256, 0, stream>>>(tbf, router_w, W_seg, wdense, seg);

    // 11) shared expert: g, u, act
    gemm_wmma_kernel<<<dim3(VAL_DIM/128, SEQ/64), 256, 0, stream>>>(
        tbf, wsg_bf, gbuf, SEQ, VAL_DIM, DIM, nullptr, 0, 0);
    gemm_wmma_kernel<<<dim3(VAL_DIM/128, SEQ/64), 256, 0, stream>>>(
        tbf, wsu_bf, ubuf, SEQ, VAL_DIM, DIM, nullptr, 0, 0);
    silu_mul_kernel<<<(SEQ * VAL_DIM) / 256, 256, 0, stream>>>(
        gbuf, ubuf, abf, SEQ * VAL_DIM);

    // 12) init output x region with x1, then accumulate MoE contributions
    hipMemcpyAsync(outX, x1, (size_t)SEQ * DIM * sizeof(float),
                   hipMemcpyDeviceToDevice, stream);

    // shared-expert down, row-scaled by sigmoid gate, accumulated into out
    gemm_wmma_kernel<<<dim3(DIM/128, SEQ/64), 256, 0, stream>>>(
        abf, wsd_bf, outX, SEQ, DIM, VAL_DIM, seg, 1, 1);

    // 13) routed experts (dense over all 8, weighted by wdense — matches ref)
    for (int e = 0; e < NE; ++e) {
        const unsigned short* Wgu_e = wgu_bf + (size_t)e * (2 * IMOE) * DIM;  // [2I][D]
        const unsigned short* Wdn_e = wdn_bf + (size_t)e * DIM * IMOE;        // [D][I]
        gemm_wmma_kernel<<<dim3((2 * IMOE)/128, SEQ/64), 256, 0, stream>>>(
            tbf, Wgu_e, gbuf, SEQ, 2 * IMOE, DIM, nullptr, 0, 0);
        glu_kernel<<<(SEQ * IMOE) / 256, 256, 0, stream>>>(gbuf, abf);
        gemm_wmma_kernel<<<dim3(DIM/128, SEQ/64), 256, 0, stream>>>(
            abf, Wdn_e, outX, SEQ, DIM, IMOE, wdense + e, NE, 1);
    }
}
